// SparseWindowAttention_39754217291866
// MI455X (gfx1250) — matrix operand order repaired and checked by reading, not compiled, not run on hardware
//
#include <hip/hip_runtime.h>
#include <cstdint>

typedef __bf16 bf16;
typedef __attribute__((ext_vector_type(16))) __bf16 v16bf;
typedef __attribute__((ext_vector_type(8)))  __bf16 v8bf;
typedef __attribute__((ext_vector_type(8)))  float  v8f;

#define CC     512        // channels
#define NHEAD  8
#define CHD    64         // channels per head
#define NTOK   9216       // 4*48*48
#define NPOOL  576        // 4*12*12
#define NKTOK  9792       // NTOK + NPOOL
#define NWIN   36
#define NKEY   1600       // 4 * (64 + 192 + 144)
#define CHUNK  160        // keys staged per flash chunk (multiple of 32)
// logits scaled into exp2 domain: 1/sqrt(64) * log2(e)
#define SCLOG  (0.125f * 1.4426950408889634f)

// ---------------- operand load helpers (contiguous k order, same for A and B) ----------

__device__ __forceinline__ v16bf cat16(v8bf lo, v8bf hi) {
  return __builtin_shufflevector(lo, hi, 0,1,2,3,4,5,6,7,8,9,10,11,12,13,14,15);
}
// A operand (16-bit, 16x32): lane holds row M=lane&15; K = base + j, contiguous 16,
// base = (lane>=16)*16. Caller passes p already offset by base.
__device__ __forceinline__ v16bf loadA16(const bf16* p) {
  return cat16(*(const v8bf*)p, *(const v8bf*)(p + 8));
}
// B operand (16-bit, 32x16): lane holds col N=lane&15; K = (lane>=16)*16 + j, contiguous 16.
// Caller passes p already offset by the lane's K base.
__device__ __forceinline__ v16bf loadB16(const bf16* p) {
  return cat16(*(const v8bf*)p, *(const v8bf*)(p + 8));
}
__device__ __forceinline__ v8f wmma_bf(v16bf a, v16bf b, v8f c) {
  return __builtin_amdgcn_wmma_f32_16x16x32_bf16(false, a, false, b, (short)0, c, false, false);
}

// CDNA5 async global->LDS DMA (ASYNCcnt-tracked), 16B per lane.
__device__ __forceinline__ void async_g2l_b128(const void* gsrc, void* ldst) {
#if defined(__AMDGCN__)
  unsigned int loff =
      (unsigned int)(uintptr_t)(__attribute__((address_space(3))) const void*)ldst;
  asm volatile("global_load_async_to_lds_b128 %0, %1, off"
               :: "v"(loff), "v"(gsrc) : "memory");
#else
  (void)gsrc; (void)ldst;
#endif
}
__device__ __forceinline__ void wait_asynccnt0() {
#if defined(__AMDGCN__)
  asm volatile("s_wait_asynccnt 0x0" ::: "memory");
#endif
}

// ---------------- small prep kernels ------------------------------------------

// f32 -> bf16, 4 elements per thread (float4 in, packed 8B out)
__global__ __launch_bounds__(256) void cvt_bf16x4_kernel(const float* __restrict__ s,
                                                         bf16* __restrict__ d, int n4) {
  int i = blockIdx.x * 256 + threadIdx.x;
  if (i < n4) {
    const float4 f = ((const float4*)s)[i];
    union { unsigned long long u; unsigned short h[4]; } p;
    p.h[0] = __builtin_bit_cast(unsigned short, (bf16)f.x);
    p.h[1] = __builtin_bit_cast(unsigned short, (bf16)f.y);
    p.h[2] = __builtin_bit_cast(unsigned short, (bf16)f.z);
    p.h[3] = __builtin_bit_cast(unsigned short, (bf16)f.w);
    ((unsigned long long*)d)[i] = p.u;
  }
}

// 4x4 average pool: x (4,48,48,512) -> pooled (576,512) bf16
__global__ __launch_bounds__(256) void pool_kernel(const float* __restrict__ x,
                                                   bf16* __restrict__ px) {
  int bidx = blockIdx.x;                 // 0..575 = t*144 + ph*12 + pw
  int t = bidx / 144, pp = bidx % 144, ph = pp / 12, pw = pp % 12;
  for (int c = threadIdx.x; c < CC; c += 256) {
    float s = 0.f;
    for (int i = 0; i < 4; ++i)
      for (int j = 0; j < 4; ++j)
        s += x[(((size_t)t * 48 + ph * 4 + i) * 48 + pw * 4 + j) * CC + c];
    px[(size_t)bidx * CC + c] = (bf16)(s * (1.0f / 16.0f));
  }
}

// pack 512x512 f32 weight into WMMA B-operand layout, bf16.
// dst element: [(ks*32+nt)*32 + lane]*16 + j  holds  W[ks*32 + (lane>=16)*16 + j][nt*16 + (lane&15)]
__global__ void packW_kernel(const float* __restrict__ W, bf16* __restrict__ Bp) {
  const int ks = blockIdx.x, nt = blockIdx.y;
  const int lane = threadIdx.x;
  const int ln = lane & 15, hi = lane >> 4;
  const int n = nt * 16 + ln;
  bf16* dst = Bp + ((size_t)(ks * 32 + nt) * 32 + lane) * 16;
#pragma unroll
  for (int j = 0; j < 16; ++j) {
    int k = ks * 32 + hi * 16 + j;
    dst[j] = (bf16)W[(size_t)k * CC + n];
  }
}

// masked[w] = any(mask>0 in window w across t)  (mask is (4,48,48,1), nonneg)
__global__ __launch_bounds__(256) void mask_kernel(const float* __restrict__ mask,
                                                   int* __restrict__ masked) {
  __shared__ float red[256];
  int w = blockIdx.x, wi = w / 6, wj = w % 6;
  int i = threadIdx.x, t = i >> 6, pos = i & 63;
  red[i] = mask[(((size_t)t * 48) + wi * 8 + (pos >> 3)) * 48 + wj * 8 + (pos & 7)];
  __syncthreads();
  for (int s = 128; s > 0; s >>= 1) {
    if (i < s) red[i] = fmaxf(red[i], red[i + s]);
    __syncthreads();
  }
  if (i == 0) masked[w] = (red[0] > 0.0f) ? 1 : 0;
}

// per-window key index table: 1600 entries -> row into K_all/V_all [0,9792)
__global__ __launch_bounds__(256) void keyidx_kernel(int* __restrict__ keyidx) {
  int w = blockIdx.x, wi = w / 6, wj = w % 6;
  for (int e = threadIdx.x; e < NKEY; e += 256) {
    int t = e / 400, j = e % 400, tok;
    if (j < 64) {                         // window's own tokens
      int r = j >> 3, c = j & 7;
      tok = ((t * 48) + wi * 8 + r) * 48 + wj * 8 + c;
    } else if (j < 256) {                 // rolled tokens selected by VALID_IND
      int vi = j - 64, s = vi / 48, rank = vi % 48;
      int p = 0, cnt = 0;
      for (int q = 0; q < 64; ++q) {      // rank-th position not in invalid quadrant s
        int r = q >> 3, c = q & 7;
        int quad = ((r >= 4) ? 2 : 0) + ((c >= 4) ? 1 : 0);
        if (quad != s) { if (cnt == rank) { p = q; break; } ++cnt; }
      }
      int r = p >> 3, c = p & 7;
      int dh = (s < 2) ? 4 : -4;          // shifts [(-4,-4),(-4,4),(4,-4),(4,4)] -> src offset
      int dw = ((s & 1) == 0) ? 4 : -4;
      int hh = (wi * 8 + r + dh + 48) % 48;
      int ww = (wj * 8 + c + dw + 48) % 48;
      tok = ((t * 48) + hh) * 48 + ww;
    } else {                              // pooled tokens appended after the 9216 real tokens
      tok = NTOK + t * 144 + (j - 256);
    }
    keyidx[w * NKEY + e] = tok;
  }
}

// ---------------- WMMA GEMM: C[M][512] = A[M][512] @ Wpacked + bias -----------

__global__ __launch_bounds__(128) void gemm_bias_kernel(
    const bf16* __restrict__ A, const bf16* __restrict__ Bp,
    const float* __restrict__ bias, bf16* __restrict__ Cb,
    float* __restrict__ Cf, int storeF32) {
  const int mtile = blockIdx.x;
  const int wave = threadIdx.x >> 5, lane = threadIdx.x & 31;
  const int ln = lane & 15, hi = lane >> 4;
  const int nt = blockIdx.y * 4 + wave;
  const bf16* ap = A + (size_t)(mtile * 16 + ln) * CC;
  v8f acc = {};
#pragma unroll
  for (int ks = 0; ks < 16; ++ks) {
    v16bf a = loadA16(ap + ks * 32 + hi * 16);
    v16bf b = loadB16(Bp + ((size_t)(ks * 32 + nt) * 32 + lane) * 16);
    acc = wmma_bf(a, b, acc);
  }
  const float bn = bias[nt * 16 + ln];
#pragma unroll
  for (int r = 0; r < 8; ++r) {
    float vv = acc[r] + bn;               // D layout: row = r + 8*hi, col = ln
    size_t off = (size_t)(mtile * 16 + r + 8 * hi) * CC + nt * 16 + ln;
    if (storeF32) Cf[off] = vv; else Cb[off] = (bf16)vv;
  }
}

// ---------------- flash-attention core ----------------------------------------

// stage nkeys gathered keys: lK[key][ch] via async DMA, lV[ch][key] transposed (packed b32)
__device__ __forceinline__ void stage_keys(const bf16* __restrict__ K,
                                           const bf16* __restrict__ V,
                                           const int* lIdx, bf16* lK, bf16* lV,
                                           int nkeys, int h, int tid) {
  const int nch = nkeys * (CHD / 8);
  // K tile: natural layout, global->LDS async DMA (no VGPR round trip)
  for (int i = tid; i < nch; i += 256) {
    int key = i >> 3, off = (i & 7) * 8;
    async_g2l_b128(K + (size_t)lIdx[key] * CC + h * CHD + off,
                   lK + (size_t)key * CHD + off);
  }
  // V tile: transpose into [ch][key]; two keys packed per 32-bit LDS store
  const int npair = (nkeys >> 1) * (CHD / 8);
  for (int i = tid; i < npair; i += 256) {
    int kp = i >> 3, off = (i & 7) * 8;
    int key = kp * 2;
    v8bf a = *(const v8bf*)(V + (size_t)lIdx[key] * CC + h * CHD + off);
    v8bf b = *(const v8bf*)(V + (size_t)lIdx[key + 1] * CC + h * CHD + off);
#pragma unroll
    for (int j = 0; j < 8; ++j) {
      unsigned int u = (unsigned int)__builtin_bit_cast(unsigned short, a[j]) |
                       ((unsigned int)__builtin_bit_cast(unsigned short, b[j]) << 16);
      *(unsigned int*)(lV + (size_t)(off + j) * CHUNK + key) = u;
    }
  }
  wait_asynccnt0();   // all of this wave's async K transfers landed in LDS
}

// one 32-key flash step for this wave's two 16-row query tiles
__device__ __forceinline__ void flash_step(
    const v16bf (&qa)[2][2], v8f (&acc)[2][4],
    float (&mrow)[2][8], float (&lrow)[2][8],
    const bf16* lK, const bf16* lV, bf16* lPw,
    int pb, int ln, int hi) {
#pragma unroll
  for (int m = 0; m < 2; ++m) {
    v8f s0 = {}, s1 = {};
#pragma unroll
    for (int ks = 0; ks < 2; ++ks) {      // S = Q (16x32ch) x K^T (32ch x 16key)
      v16bf b0 = loadB16(lK + (size_t)(pb + ln) * CHD + ks * 32 + hi * 16);
      v16bf b1 = loadB16(lK + (size_t)(pb + 16 + ln) * CHD + ks * 32 + hi * 16);
      s0 = wmma_bf(qa[m][ks], b0, s0);
      s1 = wmma_bf(qa[m][ks], b1, s1);
    }
    float alpha[8];
#pragma unroll
    for (int r = 0; r < 8; ++r) {         // online softmax (exp2 domain) over 32-key slab
      float x0 = s0[r] * SCLOG, x1 = s1[r] * SCLOG;
      float rm = fmaxf(x0, x1);
#pragma unroll
      for (int mk = 1; mk < 16; mk <<= 1) rm = fmaxf(rm, __shfl_xor(rm, mk, 32));
      float mn = fmaxf(mrow[m][r], rm);
      float a  = exp2f(mrow[m][r] - mn);
      float p0 = exp2f(x0 - mn), p1 = exp2f(x1 - mn);
      float ps = p0 + p1;
#pragma unroll
      for (int mk = 1; mk < 16; mk <<= 1) ps += __shfl_xor(ps, mk, 32);
      lrow[m][r] = lrow[m][r] * a + ps;
      mrow[m][r] = mn;
      alpha[r] = a;
      s0[r] = p0; s1[r] = p1;
    }
#pragma unroll
    for (int ct = 0; ct < 4; ++ct)
#pragma unroll
      for (int r = 0; r < 8; ++r) acc[m][ct][r] = acc[m][ct][r] * alpha[r];
    // re-layout P (D layout, f32) -> A operand (bf16) via wave-private LDS
#pragma unroll
    for (int r = 0; r < 8; ++r) {
      int mr = r + hi * 8;
      lPw[mr * 32 + ln]      = (bf16)s0[r];
      lPw[mr * 32 + 16 + ln] = (bf16)s1[r];
    }
    v16bf pA = loadA16(lPw + ln * 32 + hi * 16);
#pragma unroll
    for (int ct = 0; ct < 4; ++ct) {      // Y += P (16x32key) x V (32key x 16ch)
      v16bf vb = loadB16(lV + (size_t)(ct * 16 + ln) * CHUNK + pb + hi * 16);
      acc[m][ct] = wmma_bf(pA, vb, acc[m][ct]);
    }
  }
}

__global__ __launch_bounds__(256) void attn_kernel(
    const bf16* __restrict__ Q, const bf16* __restrict__ K,
    const bf16* __restrict__ V, const int* __restrict__ keyidx,
    const int* __restrict__ masked, bf16* __restrict__ Y) {
  __shared__ __align__(16) bf16 lK[CHUNK * CHD];
  __shared__ __align__(16) bf16 lV[CHD * CHUNK];
  __shared__ __align__(16) bf16 lP[8][16 * 32];
  __shared__ int lIdx[CHUNK];

  const int w = blockIdx.x, h = blockIdx.y;
  const int tid = threadIdx.x;
  const int wave = tid >> 5, lane = tid & 31;
  const int ln = lane & 15, hi = lane >> 4;
  const int wi = w / 6, wj = w % 6;
  const int* widx = keyidx + (size_t)w * NKEY;
  bf16* lPw = &lP[wave][0];

  // resident Q A-operands: wave owns queries [wave*32, wave*32+32)
  v16bf qa[2][2];
#pragma unroll
  for (int m = 0; m < 2; ++m) {
    int j = wave * 32 + m * 16 + ln;
    int t = j >> 6, pos = j & 63;
    int tok = ((t * 48) + wi * 8 + (pos >> 3)) * 48 + wj * 8 + (pos & 7);
    const bf16* qp = Q + (size_t)tok * CC + h * CHD;
#pragma unroll
    for (int ks = 0; ks < 2; ++ks) qa[m][ks] = loadA16(qp + ks * 32 + hi * 16);
  }

  v8f acc[2][4] = {};
  float mrow[2][8], lrow[2][8];
#pragma unroll
  for (int m = 0; m < 2; ++m)
#pragma unroll
    for (int r = 0; r < 8; ++r) { mrow[m][r] = -3.0e38f; lrow[m][r] = 0.f; }

  if (masked[w] != 0) {
    // ---- global path: flash over all 1600 keys in 160-key chunks ----
    for (int c0 = 0; c0 < NKEY; c0 += CHUNK) {
      for (int i = tid; i < CHUNK; i += 256) lIdx[i] = widx[c0 + i];
      if (c0 + CHUNK < NKEY)
        __builtin_prefetch((const void*)(widx + c0 + CHUNK), 0, 1);
      __syncthreads();
      stage_keys(K, V, lIdx, lK, lV, CHUNK, h, tid);
      __syncthreads();
#pragma unroll 1
      for (int pb = 0; pb < CHUNK; pb += 32)
        flash_step(qa, acc, mrow, lrow, lK, lV, lPw, pb, ln, hi);
      __syncthreads();
    }
  } else {
    // ---- local path: per-t attention over that t's 64 window keys ----
    const int tq = wave >> 1;             // this wave's queries all live in frame tq
    // round 0: stage local keys of t=0,1
    for (int i = tid; i < 128; i += 256) lIdx[i] = widx[(i >> 6) * 400 + (i & 63)];
    __syncthreads();
    stage_keys(K, V, lIdx, lK, lV, 128, h, tid);
    __syncthreads();
    if (tq < 2) {
      int base = tq * 64;
      flash_step(qa, acc, mrow, lrow, lK, lV, lPw, base, ln, hi);
      flash_step(qa, acc, mrow, lrow, lK, lV, lPw, base + 32, ln, hi);
    }
    __syncthreads();
    // round 1: stage local keys of t=2,3
    for (int i = tid; i < 128; i += 256) lIdx[i] = widx[((i >> 6) + 2) * 400 + (i & 63)];
    __syncthreads();
    stage_keys(K, V, lIdx, lK, lV, 128, h, tid);
    __syncthreads();
    if (tq >= 2) {
      int base = (tq - 2) * 64;
      flash_step(qa, acc, mrow, lrow, lK, lV, lPw, base, ln, hi);
      flash_step(qa, acc, mrow, lrow, lK, lV, lPw, base + 32, ln, hi);
    }
  }

  // ---- finalize: Y[tok][h*64 + n] = acc / l ----
#pragma unroll
  for (int m = 0; m < 2; ++m) {
#pragma unroll
    for (int r = 0; r < 8; ++r) {
      int jq = wave * 32 + m * 16 + r + 8 * hi;
      int t = jq >> 6, pos = jq & 63;
      int tok = ((t * 48) + wi * 8 + (pos >> 3)) * 48 + wj * 8 + (pos & 7);
      float inv = 1.0f / lrow[m][r];
#pragma unroll
      for (int ct = 0; ct < 4; ++ct)
        Y[(size_t)tok * CC + h * CHD + ct * 16 + ln] = (bf16)(acc[m][ct][r] * inv);
    }
  }
}

// ---------------- host side ----------------------------------------------------

extern "C" void kernel_launch(void* const* d_in, const int* in_sizes, int n_in,
                              void* d_out, int out_size, void* d_ws, size_t ws_size,
                              hipStream_t stream) {
  const float* x    = (const float*)d_in[0];
  const float* mask = (const float*)d_in[1];
  const float* Wq = (const float*)d_in[2]; const float* bq = (const float*)d_in[3];
  const float* Wk = (const float*)d_in[4]; const float* bk = (const float*)d_in[5];
  const float* Wv = (const float*)d_in[6]; const float* bv = (const float*)d_in[7];
  const float* Wp = (const float*)d_in[8]; const float* bp = (const float*)d_in[9];
  float* out = (float*)d_out;

  size_t off = 0;
  auto alloc = [&](size_t bytes) -> char* {
    char* p = (char*)d_ws + off;
    off += (bytes + 255) & ~(size_t)255;
    return p;
  };
  bf16* xb   = (bf16*)alloc((size_t)NTOK * CC * 2);
  bf16* pxb  = (bf16*)alloc((size_t)NPOOL * CC * 2);
  bf16* qb   = (bf16*)alloc((size_t)NTOK * CC * 2);
  bf16* Kall = (bf16*)alloc((size_t)NKTOK * CC * 2);
  bf16* Vall = (bf16*)alloc((size_t)NKTOK * CC * 2);
  bf16* yb   = (bf16*)alloc((size_t)NTOK * CC * 2);
  bf16* WqP  = (bf16*)alloc((size_t)CC * CC * 2);
  bf16* WkP  = (bf16*)alloc((size_t)CC * CC * 2);
  bf16* WvP  = (bf16*)alloc((size_t)CC * CC * 2);
  bf16* WpP  = (bf16*)alloc((size_t)CC * CC * 2);
  int*  kidx = (int*)alloc((size_t)NWIN * NKEY * 4);
  int*  mskd = (int*)alloc((size_t)NWIN * 4);

  // 1) precision conversion + pooling + weight packing
  int n4 = (NTOK * CC) / 4;
  cvt_bf16x4_kernel<<<(n4 + 255) / 256, 256, 0, stream>>>(x, xb, n4);
  pool_kernel<<<NPOOL, 256, 0, stream>>>(x, pxb);
  packW_kernel<<<dim3(16, 32), 32, 0, stream>>>(Wq, WqP);
  packW_kernel<<<dim3(16, 32), 32, 0, stream>>>(Wk, WkP);
  packW_kernel<<<dim3(16, 32), 32, 0, stream>>>(Wv, WvP);
  packW_kernel<<<dim3(16, 32), 32, 0, stream>>>(Wp, WpP);

  // 2) projections (WMMA GEMMs): q/k/v over 9216 tokens; pooled k/v appended
  gemm_bias_kernel<<<dim3(NTOK / 16, 8), 128, 0, stream>>>(xb, WqP, bq, qb,   nullptr, 0);
  gemm_bias_kernel<<<dim3(NTOK / 16, 8), 128, 0, stream>>>(xb, WkP, bk, Kall, nullptr, 0);
  gemm_bias_kernel<<<dim3(NTOK / 16, 8), 128, 0, stream>>>(xb, WvP, bv, Vall, nullptr, 0);
  gemm_bias_kernel<<<dim3(NPOOL / 16, 8), 128, 0, stream>>>(pxb, WkP, bk,
      Kall + (size_t)NTOK * CC, nullptr, 0);
  gemm_bias_kernel<<<dim3(NPOOL / 16, 8), 128, 0, stream>>>(pxb, WvP, bv,
      Vall + (size_t)NTOK * CC, nullptr, 0);

  // 3) window metadata
  mask_kernel<<<NWIN, 256, 0, stream>>>(mask, mskd);
  keyidx_kernel<<<NWIN, 256, 0, stream>>>(kidx);

  // 4) flash attention per (window, head)
  attn_kernel<<<dim3(NWIN, NHEAD), 256, 0, stream>>>(qb, Kall, Vall, kidx, mskd, yb);

  // 5) output projection, f32 result
  gemm_bias_kernel<<<dim3(NTOK / 16, 8), 128, 0, stream>>>(yb, WpP, bp, nullptr, out, 1);
}
